// AttentionClassificationHead_88656714924477
// MI455X (gfx1250) — compile-verified, matrix operand order checked by reading, not run on hardware
//
#include <hip/hip_runtime.h>
#include <hip/hip_bf16.h>
#include <cstdint>
#include <cstddef>

typedef __attribute__((ext_vector_type(16))) __bf16         v16bf;
typedef __attribute__((ext_vector_type(16))) unsigned short v16u;
typedef __attribute__((ext_vector_type(8)))  float          v8f;

static __device__ __forceinline__ unsigned short f2bf(float f) {
  unsigned int u = __builtin_bit_cast(unsigned int, f);
  u += 0x7FFFu + ((u >> 16) & 1u);     // round-to-nearest-even
  return (unsigned short)(u >> 16);
}

// ---------------- conversion / transpose kernels ----------------

__global__ void k_xt(const float* __restrict__ x, unsigned short* __restrict__ xt) {
  // x: [128][1024][196] f32 -> xt: [128*196][1024] bf16
  int i = blockIdx.x * blockDim.x + threadIdx.x;
  const int total = 128 * 1024 * 196;
  if (i >= total) return;
  int l = i % 196;
  int c = (i / 196) % 1024;
  int b = i / (196 * 1024);
  xt[((size_t)(b * 196 + l)) * 1024 + c] = f2bf(x[i]);
}

__global__ void k_wt(const float* __restrict__ w, unsigned short* __restrict__ wt, int K, int N) {
  // w: [K][N] f32 -> wt: [N][K] bf16 (transpose so B-fragments read contiguous K)
  int i = blockIdx.x * blockDim.x + threadIdx.x;
  if (i >= K * N) return;
  int n = i % N, k = i / N;
  wt[(size_t)n * K + k] = f2bf(w[i]);
}

// ---------------- generic bf16 WMMA GEMM: C = epi(A @ Wt^T) ----------------
// A: [M][K] bf16 row-major, Bt: [N][K] bf16 row-major (weight transposed).
// block = 256 threads = 8 waves; tile 64(M) x 64(N); staged K = 64 (2 WMMA K-steps,
// 4 v_wmma per barrier pair).

template<int ACT, bool BIAS, bool RES, bool OUTF, bool OUTB, bool SCALEQ>
__global__ __launch_bounds__(256) void k_gemm(
    const unsigned short* __restrict__ A,
    const unsigned short* __restrict__ Bt,
    const float* __restrict__ bias,
    const float* __restrict__ Res,
    float* __restrict__ Cf,
    unsigned short* __restrict__ Cb,
    int M, int N, int K)
{
  constexpr int LK = 72;                       // padded LDS K-stride (bf16 elems)
  __shared__ unsigned short As[64 * LK];
  __shared__ unsigned short Bs[64 * LK];
  const int tid  = threadIdx.x;
  const int lane = tid & 31;
  const int wave = tid >> 5;
  const int wm = wave & 3;                     // 16-row slab
  const int wn = wave >> 2;                    // 32-col slab
  const int bm = blockIdx.x * 64;
  const int bn = blockIdx.y * 64;
  const int lrow = tid >> 2;                   // 0..63
  const int lcol = (tid & 3) * 16;             // 0,16,32,48
  const int kb = (lane < 16) ? 0 : 8;          // per-lane K phase (ISA 16-bit A/B layout)
  const int hl = lane & 15;

  v8f acc0 = {}, acc1 = {};

  for (int k0 = 0; k0 < K; k0 += 64) {
    const unsigned short* ga = A  + (size_t)(bm + lrow) * K + k0 + lcol;
    const unsigned short* gb = Bt + (size_t)(bn + lrow) * K + k0 + lcol;
    uint4 da0 = *(const uint4*)(ga);
    uint4 da1 = *(const uint4*)(ga + 8);
    uint4 db0 = *(const uint4*)(gb);
    uint4 db1 = *(const uint4*)(gb + 8);
    *(uint4*)(&As[lrow * LK + lcol])     = da0;
    *(uint4*)(&As[lrow * LK + lcol + 8]) = da1;
    *(uint4*)(&Bs[lrow * LK + lcol])     = db0;
    *(uint4*)(&Bs[lrow * LK + lcol + 8]) = db1;
    __syncthreads();

#pragma unroll
    for (int kk = 0; kk < 64; kk += 32) {
      v16u au, bu0, bu1;
      const int ar  = (wm * 16 + hl) * LK + kk + kb;
      const int br0 = (wn * 32 + hl) * LK + kk + kb;
      const int br1 = br0 + 16 * LK;
#pragma unroll
      for (int i = 0; i < 8; ++i) {
        au[i]  = As[ar + i];    au[i + 8]  = As[ar + 16 + i];
        bu0[i] = Bs[br0 + i];   bu0[i + 8] = Bs[br0 + 16 + i];
        bu1[i] = Bs[br1 + i];   bu1[i + 8] = Bs[br1 + 16 + i];
      }
      v16bf a  = __builtin_bit_cast(v16bf, au);
      v16bf b0 = __builtin_bit_cast(v16bf, bu0);
      v16bf b1 = __builtin_bit_cast(v16bf, bu1);
      acc0 = __builtin_amdgcn_wmma_f32_16x16x32_bf16(false, a, false, b0, (short)0, acc0, false, false);
      acc1 = __builtin_amdgcn_wmma_f32_16x16x32_bf16(false, a, false, b1, (short)0, acc1, false, false);
    }
    __syncthreads();
  }

  const int ml = (lane < 16) ? 0 : 8;
#pragma unroll
  for (int r = 0; r < 8; ++r) {
    const int gm = bm + wm * 16 + r + ml;
#pragma unroll
    for (int s = 0; s < 2; ++s) {
      const int gn = bn + wn * 32 + s * 16 + hl;
      float c = (s == 0) ? acc0[r] : acc1[r];
      if constexpr (SCALEQ) { if (gn < 384) c *= 0.17677669529663687f; }  // HD^-0.5
      if constexpr (BIAS) c += bias[gn];
      if constexpr (ACT == 1) c = fminf(fmaxf(c, 0.f), 6.f);              // relu6
      if constexpr (RES) c += Res[(size_t)gm * N + gn];
      if constexpr (OUTF) Cf[(size_t)gm * N + gn] = c;
      if constexpr (OUTB) Cb[(size_t)gm * N + gn] = f2bf(c);
    }
  }
}

// ---------------- depthwise 3x3 conv + residual ----------------

__global__ void k_conv(const float* __restrict__ T0, const float* __restrict__ cw,
                       const float* __restrict__ cb, float* __restrict__ T1) {
  int i = blockIdx.x * blockDim.x + threadIdx.x;
  const int total = 128 * 196 * 384;
  if (i >= total) return;
  int d = i % 384;
  int l = (i / 384) % 196;
  int b = i / (384 * 196);
  int h = l / 14, w = l % 14;
  float s = cb[d];
#pragma unroll
  for (int kh = 0; kh < 3; ++kh)
#pragma unroll
    for (int kw = 0; kw < 3; ++kw) {
      int hh = h + kh - 1, ww = w + kw - 1;
      if (hh >= 0 && hh < 14 && ww >= 0 && ww < 14)
        s += T0[((size_t)b * 196 + hh * 14 + ww) * 384 + d] * cw[d * 9 + kh * 3 + kw];
    }
  T1[i] = s + T0[i];
}

// ---------------- LayerNorm over D=384 -> bf16 ----------------

__global__ __launch_bounds__(128) void k_ln(const float* __restrict__ X, const float* __restrict__ g,
                                            const float* __restrict__ be, unsigned short* __restrict__ Y) {
  __shared__ float ws1[4], ws2[4];
  const int row = blockIdx.x;
  const int tid = threadIdx.x, lane = tid & 31, wave = tid >> 5;
  const float* x = X + (size_t)row * 384;
  float s = 0.f, s2 = 0.f;
  for (int c = tid; c < 384; c += 128) { float v = x[c]; s += v; s2 += v * v; }
  for (int o = 16; o; o >>= 1) { s += __shfl_xor(s, o, 32); s2 += __shfl_xor(s2, o, 32); }
  if (lane == 0) { ws1[wave] = s; ws2[wave] = s2; }
  __syncthreads();
  s  = ws1[0] + ws1[1] + ws1[2] + ws1[3];
  s2 = ws2[0] + ws2[1] + ws2[2] + ws2[3];
  float mu  = s * (1.f / 384.f);
  float var = s2 * (1.f / 384.f) - mu * mu;
  float inv = rsqrtf(var + 1e-5f);
  for (int c = tid; c < 384; c += 128)
    Y[(size_t)row * 384 + c] = f2bf((x[c] - mu) * inv * g[c] + be[c]);
}

// ---------------- fused attention with contextual RPE ----------------
// One workgroup (128 thr = 4 waves) per (b, h, 16-query tile). L=196 padded to 208/224.

__global__ __launch_bounds__(128) void k_attn(
    const unsigned short* __restrict__ qkv,   // [B][196][1152] bf16, q pre-scaled
    const float* __restrict__ rpe,            // [49][32] f32
    const int* __restrict__ bucket,           // [196][196]
    unsigned short* __restrict__ O)           // [B][196][384] bf16
{
  constexpr int LQ = 40;
  __shared__ unsigned short Qs[16 * LQ];
  __shared__ unsigned short Ks[208 * LQ];
  __shared__ unsigned short Rs[64 * LQ];
  __shared__ unsigned short Vts[32 * 232];    // V transposed: [d][j]
  __shared__ float Ss[16 * 224];
  __shared__ float Rlog[16 * 64];
  __shared__ unsigned short Ps[16 * 232];
  __shared__ float rowsum[16];

  const int qt = blockIdx.x, h = blockIdx.y, b = blockIdx.z;
  const int tid = threadIdx.x, lane = tid & 31, wave = tid >> 5;
  const size_t base = (size_t)b * 196 * 1152;
  const int cq = h * 32, ck = 384 + h * 32, cv = 768 + h * 32;

  for (int idx = tid; idx < 16 * 32; idx += 128) {
    int r = idx >> 5, c = idx & 31;
    int i = qt * 16 + r;
    Qs[r * LQ + c] = (i < 196) ? qkv[base + (size_t)i * 1152 + cq + c] : (unsigned short)0;
  }
  for (int j = tid; j < 208; j += 128)
    for (int c = 0; c < 32; ++c)
      Ks[j * LQ + c] = (j < 196) ? qkv[base + (size_t)j * 1152 + ck + c] : (unsigned short)0;
  for (int j = tid; j < 224; j += 128)
    for (int c = 0; c < 32; ++c)
      Vts[c * 232 + j] = (j < 196) ? qkv[base + (size_t)j * 1152 + cv + c] : (unsigned short)0;
  for (int n = tid; n < 64; n += 128)
    for (int c = 0; c < 32; ++c)
      Rs[n * LQ + c] = (n < 49) ? f2bf(rpe[n * 32 + c]) : (unsigned short)0;
  __syncthreads();

  const int kb = (lane < 16) ? 0 : 8;
  const int hl = lane & 15;
  const int ml = (lane < 16) ? 0 : 8;

  // Q fragment (A matrix, K=32): one per wave, reused across column tiles
  v16u qu;
#pragma unroll
  for (int i = 0; i < 8; ++i) { qu[i] = Qs[hl * LQ + kb + i]; qu[i + 8] = Qs[hl * LQ + kb + 16 + i]; }
  v16bf qa = __builtin_bit_cast(v16bf, qu);

  // RPE logits: wave w owns exactly the 16-col tile w (4 tiles, 4 waves)
  {
    const unsigned short* Bsrc = &Rs[(wave * 16) * LQ];
    v16u bu;
#pragma unroll
    for (int i = 0; i < 8; ++i) { bu[i] = Bsrc[hl * LQ + kb + i]; bu[i + 8] = Bsrc[hl * LQ + kb + 16 + i]; }
    v16bf bb = __builtin_bit_cast(v16bf, bu);
    v8f acc = {};
    acc = __builtin_amdgcn_wmma_f32_16x16x32_bf16(false, qa, false, bb, (short)0, acc, false, false);
#pragma unroll
    for (int r = 0; r < 8; ++r)
      Rlog[(r + ml) * 64 + wave * 16 + hl] = acc[r];
  }

  // S = Q K^T : 13 column tiles strided over the 4 waves
  for (int ct = wave; ct < 13; ct += 4) {
    const unsigned short* Bsrc = &Ks[(ct * 16) * LQ];
    v16u bu;
#pragma unroll
    for (int i = 0; i < 8; ++i) { bu[i] = Bsrc[hl * LQ + kb + i]; bu[i + 8] = Bsrc[hl * LQ + kb + 16 + i]; }
    v16bf bb = __builtin_bit_cast(v16bf, bu);
    v8f acc = {};
    acc = __builtin_amdgcn_wmma_f32_16x16x32_bf16(false, qa, false, bb, (short)0, acc, false, false);
#pragma unroll
    for (int r = 0; r < 8; ++r)
      Ss[(r + ml) * 224 + ct * 16 + hl] = acc[r];
  }
  __syncthreads();

  // gather-add RPE + softmax; 8 threads per row, width-8 shuffles
  {
    const int r = tid >> 3, sub = tid & 7;
    const int qi = qt * 16 + r;
    float mx = -1e30f, sm = 0.f;
    if (qi < 196) {
      for (int j = sub; j < 196; j += 8) {
        float s = Ss[r * 224 + j] + Rlog[r * 64 + bucket[qi * 196 + j]];
        Ss[r * 224 + j] = s;
        mx = fmaxf(mx, s);
      }
      for (int o = 4; o; o >>= 1) mx = fmaxf(mx, __shfl_xor(mx, o, 8));
      for (int j = sub; j < 196; j += 8) {
        float e = __expf(Ss[r * 224 + j] - mx);
        sm += e;
        Ps[r * 232 + j] = f2bf(e);
      }
      for (int o = 4; o; o >>= 1) sm += __shfl_xor(sm, o, 8);
      for (int j = 196 + sub; j < 224; j += 8) Ps[r * 232 + j] = 0;
    } else {
      for (int j = sub; j < 224; j += 8) Ps[r * 232 + j] = 0;
      sm = 1.f;
    }
    if (sub == 0) rowsum[r] = sm;
  }
  __syncthreads();

  // O = P V : waves 0,1 own the two 16-wide d-subtiles, K = 224 (7 steps)
  if (wave < 2) {
    v8f acc = {};
    for (int ks = 0; ks < 7; ++ks) {
      const int k0 = ks * 32 + kb;
      v16u au, bu;
#pragma unroll
      for (int i = 0; i < 8; ++i) {
        au[i]     = Ps[hl * 232 + k0 + i];
        au[i + 8] = Ps[hl * 232 + k0 + 16 + i];
        bu[i]     = Vts[(wave * 16 + hl) * 232 + k0 + i];
        bu[i + 8] = Vts[(wave * 16 + hl) * 232 + k0 + 16 + i];
      }
      v16bf a = __builtin_bit_cast(v16bf, au);
      v16bf v = __builtin_bit_cast(v16bf, bu);
      acc = __builtin_amdgcn_wmma_f32_16x16x32_bf16(false, a, false, v, (short)0, acc, false, false);
    }
#pragma unroll
    for (int r = 0; r < 8; ++r) {
      const int i = qt * 16 + r + ml;
      if (i < 196) {
        float val = acc[r] / rowsum[r + ml];
        O[((size_t)b * 196 + i) * 384 + h * 32 + wave * 16 + hl] = f2bf(val);
      }
    }
  }
}

// ---------------- BatchNorm1d(196) over (b, d) ----------------

__global__ __launch_bounds__(256) void k_bn1_stats(const float* __restrict__ T3, float* __restrict__ st) {
  __shared__ float r1[8], r2[8];
  const int l = blockIdx.x, tid = threadIdx.x, lane = tid & 31, wave = tid >> 5;
  float s = 0.f, s2 = 0.f;
  for (int i = tid; i < 128 * 384; i += 256) {
    int b = i / 384, d = i % 384;
    float v = T3[((size_t)b * 196 + l) * 384 + d];
    s += v; s2 += v * v;
  }
  for (int o = 16; o; o >>= 1) { s += __shfl_xor(s, o, 32); s2 += __shfl_xor(s2, o, 32); }
  if (lane == 0) { r1[wave] = s; r2[wave] = s2; }
  __syncthreads();
  if (tid == 0) {
    float a = 0.f, a2 = 0.f;
    for (int w = 0; w < 8; ++w) { a += r1[w]; a2 += r2[w]; }
    st[l] = a; st[196 + l] = a2;
  }
}

__global__ void k_bn1_norm(const float* __restrict__ T3, const float* __restrict__ st,
                           const float* __restrict__ g, const float* __restrict__ be,
                           unsigned short* __restrict__ T4) {
  int i = blockIdx.x * blockDim.x + threadIdx.x;
  const int total = 128 * 196 * 384;
  if (i >= total) return;
  int l = (i / 384) % 196;
  const float inv_n = 1.f / (128.f * 384.f);
  float mu  = st[l] * inv_n;
  float var = st[196 + l] * inv_n - mu * mu;
  float v = (T3[i] - mu) * rsqrtf(var + 2e-5f) * g[l] + be[l];
  T4[i] = f2bf(v);   // flat [B][L*D] == [B,L,D]
}

// ---------------- BatchNorm1d(384) over batch -> final output ----------------

__global__ __launch_bounds__(128) void k_bn2(const float* __restrict__ Y0, const float* __restrict__ g,
                                             const float* __restrict__ be, float* __restrict__ out) {
  __shared__ float r1[4], r2[4];
  const int f = blockIdx.x, tid = threadIdx.x, lane = tid & 31, wave = tid >> 5;
  float v = Y0[(size_t)tid * 384 + f];
  float s = v, s2 = v * v;
  for (int o = 16; o; o >>= 1) { s += __shfl_xor(s, o, 32); s2 += __shfl_xor(s2, o, 32); }
  if (lane == 0) { r1[wave] = s; r2[wave] = s2; }
  __syncthreads();
  s  = r1[0] + r1[1] + r1[2] + r1[3];
  s2 = r2[0] + r2[1] + r2[2] + r2[3];
  float mu  = s * (1.f / 128.f);
  float var = s2 * (1.f / 128.f) - mu * mu;
  out[(size_t)tid * 384 + f] = (v - mu) * rsqrtf(var + 2e-5f) * g[f] + be[f];
}

// ---------------- host launcher ----------------

extern "C" void kernel_launch(void* const* d_in, const int* in_sizes, int n_in,
                              void* d_out, int out_size, void* d_ws, size_t ws_size,
                              hipStream_t stream) {
  (void)in_sizes; (void)n_in; (void)out_size; (void)ws_size;
  const float* x      = (const float*)d_in[0];
  const int*   bucket = (const int*)d_in[1];
  const float* we1    = (const float*)d_in[2];
  const float* be1    = (const float*)d_in[3];
  const float* we2    = (const float*)d_in[4];
  const float* be2    = (const float*)d_in[5];
  const float* conv_w = (const float*)d_in[6];
  const float* conv_b = (const float*)d_in[7];
  const float* ln1_g  = (const float*)d_in[8];
  const float* ln1_b  = (const float*)d_in[9];
  const float* ln2_g  = (const float*)d_in[10];
  const float* ln2_b  = (const float*)d_in[11];
  const float* wqkv   = (const float*)d_in[12];
  const float* rpe    = (const float*)d_in[13];
  const float* wproj  = (const float*)d_in[14];
  const float* bproj  = (const float*)d_in[15];
  const float* wm1    = (const float*)d_in[16];
  const float* bm1    = (const float*)d_in[17];
  const float* wm2    = (const float*)d_in[18];
  const float* bm2    = (const float*)d_in[19];
  const float* bn1_g  = (const float*)d_in[20];
  const float* bn1_b  = (const float*)d_in[21];
  const float* bn2_g  = (const float*)d_in[22];
  const float* bn2_b  = (const float*)d_in[23];
  const float* wlin   = (const float*)d_in[24];
  const float* blin   = (const float*)d_in[25];
  float* out = (float*)d_out;

  char* p = (char*)d_ws;
  size_t off = 0;
  auto take = [&](size_t sz) { char* r = p + off; off = (off + sz + 255) & ~(size_t)255; return r; };
  const size_t M = 25088;
  unsigned short* R0   = (unsigned short*)take(M * 1536 * 2);   // XT[.,1024] then M1[.,1536]
  unsigned short* R1   = (unsigned short*)take(M * 1152 * 2);   // H1[.,1024] then QKV[.,1152]
  float*          T02  = (float*)take(M * 384 * 4);             // T0 then T2
  float*          T13  = (float*)take(M * 384 * 4);             // T1 then T3
  unsigned short* SM   = (unsigned short*)take(M * 384 * 2);    // HLN -> O -> H2 -> T4
  unsigned short* we1t = (unsigned short*)take((size_t)1024 * 1024 * 2);
  unsigned short* we2t = (unsigned short*)take((size_t)1024 * 384 * 2);
  unsigned short* wqkvt= (unsigned short*)take((size_t)384 * 1152 * 2);
  unsigned short* wprt = (unsigned short*)take((size_t)384 * 384 * 2);
  unsigned short* wm1t = (unsigned short*)take((size_t)384 * 1536 * 2);
  unsigned short* wm2t = (unsigned short*)take((size_t)1536 * 384 * 2);
  unsigned short* wlint= (unsigned short*)take((size_t)75264 * 384 * 2);
  float* st = (float*)take(2 * 196 * 4);
  float* Y0 = (float*)take(128 * 384 * 4);

  auto cdiv = [](long a, long b_) { return (int)((a + b_ - 1) / b_); };

  // weight / input conversion (bf16, transposed to [N][K])
  k_wt<<<cdiv(1024 * 1024, 256), 256, 0, stream>>>(we1, we1t, 1024, 1024);
  k_wt<<<cdiv(1024 * 384, 256), 256, 0, stream>>>(we2, we2t, 1024, 384);
  k_wt<<<cdiv(384 * 1152, 256), 256, 0, stream>>>(wqkv, wqkvt, 384, 1152);
  k_wt<<<cdiv(384 * 384, 256), 256, 0, stream>>>(wproj, wprt, 384, 384);
  k_wt<<<cdiv(384 * 1536, 256), 256, 0, stream>>>(wm1, wm1t, 384, 1536);
  k_wt<<<cdiv(1536 * 384, 256), 256, 0, stream>>>(wm2, wm2t, 1536, 384);
  k_wt<<<cdiv((long)75264 * 384, 256), 256, 0, stream>>>(wlin, wlint, 75264, 384);
  k_xt<<<cdiv((long)128 * 1024 * 196, 256), 256, 0, stream>>>(x, R0);

  // embed MLP: relu6(Xt@we1+be1)@we2+be2
  k_gemm<1, true, false, false, true, false><<<dim3(392, 16), 256, 0, stream>>>(
      R0, we1t, be1, nullptr, nullptr, R1, 25088, 1024, 1024);
  k_gemm<0, true, false, true, false, false><<<dim3(392, 6), 256, 0, stream>>>(
      R1, we2t, be2, nullptr, T02, nullptr, 25088, 384, 1024);
  // PosCNN depthwise conv + residual
  k_conv<<<cdiv((long)128 * 196 * 384, 256), 256, 0, stream>>>(T02, conv_w, conv_b, T13);
  // LN1 -> bf16
  k_ln<<<25088, 128, 0, stream>>>(T13, ln1_g, ln1_b, SM);
  // QKV (q columns pre-scaled by HD^-0.5)
  k_gemm<0, false, false, false, true, true><<<dim3(392, 18), 256, 0, stream>>>(
      SM, wqkvt, nullptr, nullptr, nullptr, R1, 25088, 1152, 384);
  // attention with contextual RPE
  k_attn<<<dim3(13, 12, 128), 128, 0, stream>>>(R1, rpe, bucket, SM);
  // proj + residual
  k_gemm<0, true, true, true, false, false><<<dim3(392, 6), 256, 0, stream>>>(
      SM, wprt, bproj, T13, T02, nullptr, 25088, 384, 384);
  // LN2
  k_ln<<<25088, 128, 0, stream>>>(T02, ln2_g, ln2_b, SM);
  // MLP: relu6(h@wm1+bm1)@wm2+bm2 + residual
  k_gemm<1, true, false, false, true, false><<<dim3(392, 24), 256, 0, stream>>>(
      SM, wm1t, bm1, nullptr, nullptr, R0, 25088, 1536, 384);
  k_gemm<0, true, true, true, false, false><<<dim3(392, 6), 256, 0, stream>>>(
      R0, wm2t, bm2, T02, T13, nullptr, 25088, 384, 1536);
  // BN1d(196) over (b, d)
  k_bn1_stats<<<196, 256, 0, stream>>>(T13, st);
  k_bn1_norm<<<cdiv((long)128 * 196 * 384, 256), 256, 0, stream>>>(T13, st, bn1_g, bn1_b, SM);
  // flatten + head linear
  k_gemm<0, true, false, true, false, false><<<dim3(2, 6), 256, 0, stream>>>(
      SM, wlint, blin, nullptr, Y0, nullptr, 128, 384, 75264);
  // BN1d(384) over batch -> output
  k_bn2<<<384, 128, 0, stream>>>(Y0, bn2_g, bn2_b, out);
}
